// SPD_64441689309275
// MI455X (gfx1250) — matrix operand order repaired and checked by reading, not compiled, not run on hardware
//
#include <hip/hip_runtime.h>
#include <math.h>

// ---------------------------------------------------------------------------
// SPD forward for MI455X (gfx1250, wave32, WMMA).
// All 1x1 convs run through v_wmma_f32_16x16x32_f16 (f32 -> f16 inputs,
// f32 accumulate). Epilogues (bias / BN / relu / tanh / residual / strided
// transpose-conv scatter) are fused into the GEMM. The attention 64->256->64
// conv pair over B*N*K = 262144 points is fused into one kernel with the
// 256-ch intermediate held in LDS (never hits HBM).
// Staging is branch-free (clamped addresses + selects) so no exec-mask
// serialization appears in the GEMM inner loop; tiles load via b128.
// ---------------------------------------------------------------------------

typedef __attribute__((ext_vector_type(16))) _Float16 v16h;
typedef __attribute__((ext_vector_type(8)))  float    v8f;
typedef __attribute__((ext_vector_type(4)))  unsigned int u32x4;

union FragH { u32x4 q[2]; v16h h; };

struct __attribute__((aligned(16))) F4 { float x, y, z, w; };

__device__ __forceinline__ float relu_(float v) { return v > 0.f ? v : 0.f; }
__device__ __forceinline__ int   imin_(int a, int b) { return a < b ? a : b; }

#define F_RELU_OUT 1
#define F_RELU_IN  2
#define F_BN       4
#define F_TANH     8
#define F_RES      16
#define F_RES_HALF 32

// ---------------------------------------------------------------------------
// Generic WMMA GEMM:  Y[b, o, ycol] = epilogue( sum_c W[o,c] * X[b, c, col] )
// ycol = col*colScale + colOff  (colScale=2 implements the ConvTranspose scatter)
// Workgroup: 256 threads = 8 waves; tile = 32 (O) x 64 (cols); K step 32.
// Requires M % 64 == 0 (true for all call sites: 2048/4096/32768).
// ---------------------------------------------------------------------------
__global__ __launch_bounds__(256)
void wmma_gemm(const float* __restrict__ W, const float* __restrict__ X,
               const float* __restrict__ bias,
               const float* __restrict__ bn_g, const float* __restrict__ bn_b,
               const float* __restrict__ bn_m, const float* __restrict__ bn_v,
               const float* __restrict__ res, float* __restrict__ Y,
               int O, int C, int M,
               long long strideX, int ldX,
               long long strideY, int ldY,
               long long strideR, int ldR,
               int colScale, int colOff, int flags)
{
    __shared__ _Float16 Wl[32][40];   // [o][k]   pitch 80B (16B aligned rows)
    __shared__ _Float16 Xl[64][40];   // [col][k] transposed for B-fragment

    const int b       = blockIdx.z;
    const int colBase = blockIdx.x * 64;
    const int oBase   = blockIdx.y * 32;
    const int tid  = threadIdx.x;
    const int lane = tid & 31;
    const int wave = tid >> 5;
    const int wo = wave >> 2;   // 0..1 : O-tile within workgroup
    const int wn = wave & 3;    // 0..3 : col-tile within workgroup
    const int m16 = lane & 15;
    const int lh  = lane >> 4;  // lane-half selects K subset (CDNA5 layout)

    const float* Xb = X + (long long)b * strideX;

    // staging indices (invariant across K steps)
    const int sWo  = tid >> 3;                         // 0..31 W row
    const int sWc  = (tid & 7) * 4;                    // 0..28 W col
    const int sWgo = imin_(oBase + sWo, O - 1);        // clamped row (garbage rows
                                                       // are masked at epilogue)
    const int sXc  = tid >> 3;                         // 0..31 X channel
    const int sXl  = (tid & 7) * 8;                    // 0..56 X col
    const bool doReluIn = (flags & F_RELU_IN) != 0;

    v8f acc = {0.f,0.f,0.f,0.f,0.f,0.f,0.f,0.f};

    for (int k0 = 0; k0 < C; k0 += 32) {
        const bool fullK = (k0 + 32 <= C);             // uniform scalar branch
        if (fullK) {
            // ---- fast staging: unguarded b128 loads ----
            {
                F4 w = *reinterpret_cast<const F4*>(W + (long long)sWgo * C + k0 + sWc);
                Wl[sWo][sWc+0] = (_Float16)w.x;
                Wl[sWo][sWc+1] = (_Float16)w.y;
                Wl[sWo][sWc+2] = (_Float16)w.z;
                Wl[sWo][sWc+3] = (_Float16)w.w;
            }
            {
                const float* xp = Xb + (long long)(k0 + sXc) * ldX + colBase + sXl;
                F4 a = *reinterpret_cast<const F4*>(xp);
                F4 c = *reinterpret_cast<const F4*>(xp + 4);
                float xv[8] = {a.x, a.y, a.z, a.w, c.x, c.y, c.z, c.w};
                if (doReluIn) {
#pragma unroll
                    for (int i = 0; i < 8; ++i) xv[i] = relu_(xv[i]);
                }
#pragma unroll
                for (int i = 0; i < 8; ++i) Xl[sXl + i][sXc] = (_Float16)xv[i];
            }
        } else {
            // ---- tail staging (only C%32!=0 call sites, i.e. C==3):
            //      clamped addresses + selects, no exec-masked loads ----
            {
#pragma unroll
                for (int i = 0; i < 4; ++i) {
                    int cc = k0 + sWc + i;
                    float w = W[(long long)sWgo * C + imin_(cc, C - 1)];
                    Wl[sWo][sWc + i] = (_Float16)(cc < C ? w : 0.f);
                }
            }
            {
                int gc = k0 + sXc;
                bool cvalid = gc < C;
                const float* xp = Xb + (long long)imin_(gc, C - 1) * ldX + colBase + sXl;
                F4 a = *reinterpret_cast<const F4*>(xp);
                F4 c = *reinterpret_cast<const F4*>(xp + 4);
                float xv[8] = {a.x, a.y, a.z, a.w, c.x, c.y, c.z, c.w};
#pragma unroll
                for (int i = 0; i < 8; ++i) {
                    float v = cvalid ? xv[i] : 0.f;
                    if (doReluIn) v = relu_(v);
                    Xl[sXl + i][sXc] = (_Float16)v;
                }
            }
        }
        __syncthreads();

        FragH fa, fb;
        {   // A: lane m16 = row M, lh selects K-half: elems 0..7 = K(lh*8..),
            // elems 8..15 = K(16+lh*8..)
            const _Float16* rp = &Wl[wo*16 + m16][0];
            fa.q[0] = *reinterpret_cast<const u32x4*>(rp + lh*8);
            fa.q[1] = *reinterpret_cast<const u32x4*>(rp + 16 + lh*8);
        }
        {   // B: lane m16 = column N, lh selects K-half: elems 0..7 = K(lh*8..),
            // elems 8..15 = K(16+lh*8..)
            const _Float16* cp = &Xl[wn*16 + m16][0];
            fb.q[0] = *reinterpret_cast<const u32x4*>(cp + lh*8);
            fb.q[1] = *reinterpret_cast<const u32x4*>(cp + 16 + lh*8);
        }
        acc = __builtin_amdgcn_wmma_f32_16x16x32_f16(false, fa.h, false, fb.h,
                                                     (short)0, acc, false, false);
        __syncthreads();
    }

    // ---- fused epilogue; D layout: VGPR j holds row m = j + 8*lh, col = lane&15
    const int col  = colBase + wn*16 + m16;            // always < M (64 | M)
    const int ycol = col * colScale + colOff;
#pragma unroll
    for (int j = 0; j < 8; ++j) {
        int o = oBase + wo*16 + lh*8 + j;
        if (o < O) {
            float val = acc[j];
            if (bias) val += bias[o];
            if (flags & F_BN) {
                float inv = bn_g[o] * rsqrtf(bn_v[o] + 1e-5f);
                val = (val - bn_m[o]) * inv + bn_b[o];
            }
            if (flags & F_TANH) val = tanhf(val);
            if (flags & F_RES) {
                int rc = (flags & F_RES_HALF) ? (ycol >> 1) : ycol;
                val += res[(long long)b*strideR + (long long)o*ldR + rc];
            }
            if (flags & F_RELU_OUT) val = relu_(val);
            Y[(long long)b*strideY + (long long)o*ldY + ycol] = val;
        }
    }
}

// ---------------------------------------------------------------------------
// Fused attention convs: a = W2 @ relu(BN(W1 @ x + b1)) + b2
// x: (B,64,M), mid: 256ch kept in LDS as f16, out: (B,64,M).  M = N*K = 32768.
// ---------------------------------------------------------------------------
__device__ __forceinline__ void load_afrag_global(FragH& fa, const float* wp)
{   // wp 16B-aligned; elems 0..7 = wp[0..7], elems 8..15 = wp[16..23]
    F4 a0 = *reinterpret_cast<const F4*>(wp);
    F4 a1 = *reinterpret_cast<const F4*>(wp + 4);
    F4 a2 = *reinterpret_cast<const F4*>(wp + 16);
    F4 a3 = *reinterpret_cast<const F4*>(wp + 20);
    fa.h[0]=(_Float16)a0.x;  fa.h[1]=(_Float16)a0.y;
    fa.h[2]=(_Float16)a0.z;  fa.h[3]=(_Float16)a0.w;
    fa.h[4]=(_Float16)a1.x;  fa.h[5]=(_Float16)a1.y;
    fa.h[6]=(_Float16)a1.z;  fa.h[7]=(_Float16)a1.w;
    fa.h[8]=(_Float16)a2.x;  fa.h[9]=(_Float16)a2.y;
    fa.h[10]=(_Float16)a2.z; fa.h[11]=(_Float16)a2.w;
    fa.h[12]=(_Float16)a3.x; fa.h[13]=(_Float16)a3.y;
    fa.h[14]=(_Float16)a3.z; fa.h[15]=(_Float16)a3.w;
}

__global__ __launch_bounds__(256)
void attn_fused(const float* __restrict__ W1, const float* __restrict__ b1,
                const float* __restrict__ g,  const float* __restrict__ be,
                const float* __restrict__ bm, const float* __restrict__ bv,
                const float* __restrict__ W2, const float* __restrict__ b2,
                const float* __restrict__ X,  float* __restrict__ Y, int M)
{
    __shared__ _Float16 Xl[64][72];     // [col][c]   C1 = 64
    __shared__ _Float16 midl[64][264];  // [col][o]   mid = 256 channels

    const int b       = blockIdx.y;
    const int colBase = blockIdx.x * 64;
    const int tid  = threadIdx.x;
    const int lane = tid & 31;
    const int wave = tid >> 5;
    const int wo = wave >> 2;   // 0..1
    const int wn = wave & 3;    // 0..3
    const int m16 = lane & 15;
    const int lh  = lane >> 4;

    // stage X (64 ch x 64 cols), transposed, via b128 loads
    {
        const float* Xb = X + (long long)b * 64 * M;
        int c  = tid >> 2;          // 0..63
        int i0 = (tid & 3) * 16;    // 0,16,32,48
        const float* xp = Xb + (long long)c * M + colBase + i0;
#pragma unroll
        for (int q4 = 0; q4 < 4; ++q4) {
            F4 a = *reinterpret_cast<const F4*>(xp + q4*4);
            Xl[i0 + q4*4 + 0][c] = (_Float16)a.x;
            Xl[i0 + q4*4 + 1][c] = (_Float16)a.y;
            Xl[i0 + q4*4 + 2][c] = (_Float16)a.z;
            Xl[i0 + q4*4 + 3][c] = (_Float16)a.w;
        }
    }
    __syncthreads();

    // phase 1: mid = relu(BN(W1 @ x + b1)); wave (wo,wn) owns rows wo*128..+127
    // of col group wn. A-fragments read directly from global W1 (L2 resident).
    for (int r = 0; r < 8; ++r) {
        int rt = wo*8 + r;                       // row tile 0..15
        v8f acc = {0.f,0.f,0.f,0.f,0.f,0.f,0.f,0.f};
#pragma unroll
        for (int k0 = 0; k0 < 64; k0 += 32) {
            FragH fa, fb;
            load_afrag_global(fa, W1 + (long long)(rt*16 + m16) * 64 + k0 + lh*8);
            const _Float16* cp = &Xl[wn*16 + m16][0];
            fb.q[0] = *reinterpret_cast<const u32x4*>(cp + k0 + lh*8);
            fb.q[1] = *reinterpret_cast<const u32x4*>(cp + k0 + 16 + lh*8);
            acc = __builtin_amdgcn_wmma_f32_16x16x32_f16(false, fa.h, false, fb.h,
                                                         (short)0, acc, false, false);
        }
        int col = wn*16 + m16;
#pragma unroll
        for (int j = 0; j < 8; ++j) {
            int o = rt*16 + lh*8 + j;
            float val = acc[j] + b1[o];
            float inv = g[o] * rsqrtf(bv[o] + 1e-5f);
            val = (val - bm[o]) * inv + be[o];
            midl[col][o] = (_Float16)relu_(val);
        }
    }
    __syncthreads();

    // phase 2: out = W2 @ mid + b2 (64 x 64, K=256); each wave: 2 row tiles
    v8f acc2[2];
    acc2[0] = (v8f){0.f,0.f,0.f,0.f,0.f,0.f,0.f,0.f};
    acc2[1] = (v8f){0.f,0.f,0.f,0.f,0.f,0.f,0.f,0.f};
    for (int k0 = 0; k0 < 256; k0 += 32) {
#pragma unroll
        for (int t = 0; t < 2; ++t) {
            int ot = wo*2 + t;
            FragH fa, fb;
            load_afrag_global(fa, W2 + (long long)(ot*16 + m16) * 256 + k0 + lh*8);
            const _Float16* cp = &midl[wn*16 + m16][0];
            fb.q[0] = *reinterpret_cast<const u32x4*>(cp + k0 + lh*8);
            fb.q[1] = *reinterpret_cast<const u32x4*>(cp + k0 + 16 + lh*8);
            acc2[t] = __builtin_amdgcn_wmma_f32_16x16x32_f16(false, fa.h, false, fb.h,
                                                             (short)0, acc2[t], false, false);
        }
    }
    const int col = colBase + wn*16 + m16;
    float* Yb = Y + (long long)b * 64 * M;
#pragma unroll
    for (int t = 0; t < 2; ++t) {
#pragma unroll
        for (int j = 0; j < 8; ++j) {
            int o = (wo*2 + t)*16 + lh*8 + j;
            Yb[(long long)o * M + col] = acc2[t][j] + b2[o];
        }
    }
}

// ---------------------------------------------------------------------------
// KNN: one thread per query, batch positions in LDS, register top-16.
// grid = (N/256, B)
// ---------------------------------------------------------------------------
__global__ __launch_bounds__(256)
void knn_kernel(const float* __restrict__ pos, int* __restrict__ idx, int N)
{
    __shared__ float px[2048], py[2048], pz[2048];
    const int b = blockIdx.y;
    const float* pb = pos + (long long)b * 3 * N;
    for (int i = threadIdx.x; i < N; i += 256) {
        px[i] = pb[i];
        py[i] = pb[N + i];
        pz[i] = pb[2*N + i];
    }
    __syncthreads();
    const int n = blockIdx.x * 256 + threadIdx.x;
    const float qx = px[n], qy = py[n], qz = pz[n];
    float bd[16]; int bi[16];
#pragma unroll
    for (int j = 0; j < 16; ++j) { bd[j] = 3.4e38f; bi[j] = 0; }
    for (int m = 0; m < N; ++m) {
        float dx = qx - px[m], dy = qy - py[m], dz = qz - pz[m];
        float d = dx*dx + dy*dy + dz*dz;
        if (d < bd[15]) {
            bd[15] = d; bi[15] = m;
#pragma unroll
            for (int j = 15; j > 0; --j) {       // single bubble pass keeps sorted
                if (bd[j] < bd[j-1]) {
                    float td = bd[j]; bd[j] = bd[j-1]; bd[j-1] = td;
                    int   ti = bi[j]; bi[j] = bi[j-1]; bi[j-1] = ti;
                }
            }
        }
    }
    int* op = idx + ((long long)b * N + n) * 16;
#pragma unroll
    for (int j = 0; j < 16; ++j) op[j] = bi[j];
}

// ---------------------------------------------------------------------------
// Small glue kernels
// ---------------------------------------------------------------------------
__global__ void rowmax_kernel(const float* __restrict__ f1, float* __restrict__ gmax, int N)
{   // block per (b, c in 0..127) over f1cat rows 0..127
    __shared__ float red[256];
    int c = blockIdx.x & 127;
    int b = blockIdx.x >> 7;
    const float* fp = f1 + ((long long)b * 768 + c) * N;
    float mx = -3.4e38f;
    for (int i = threadIdx.x; i < N; i += 256) mx = fmaxf(mx, fp[i]);
    red[threadIdx.x] = mx;
    __syncthreads();
    for (int s = 128; s > 0; s >>= 1) {
        if (threadIdx.x < s) red[threadIdx.x] = fmaxf(red[threadIdx.x], red[threadIdx.x + s]);
        __syncthreads();
    }
    if (threadIdx.x == 0) gmax[b*128 + c] = red[0];
}

__global__ void fill_f1cat(float* __restrict__ f1cat, const float* __restrict__ gmax,
                           const float* __restrict__ fg, int N)
{   // rows 128..255 <- gmax broadcast; rows 256..767 <- feat_global broadcast
    long long t = (long long)blockIdx.x * 256 + threadIdx.x;
    long long total = (long long)8 * 640 * N;
    if (t >= total) return;
    int n = (int)(t % N);
    int r = (int)((t / N) % 640);
    int b = (int)(t / ((long long)640 * N));
    int c = 128 + r;
    float v = (c < 256) ? gmax[b*128 + (c - 128)] : fg[b*512 + (c - 256)];
    f1cat[((long long)b*768 + c) * N + n] = v;
}

__global__ void concat128(const float* __restrict__ A, const float* __restrict__ Bv,
                          float* __restrict__ out, int N)
{   // out (B,256,N) = cat(A(B,128,N), Bv(B,128,N))
    long long t = (long long)blockIdx.x * 256 + threadIdx.x;
    long long total = (long long)8 * 256 * N;
    if (t >= total) return;
    int n = (int)(t % N);
    int c = (int)((t / N) % 256);
    int b = (int)(t / ((long long)256 * N));
    float v = (c < 128) ? A[((long long)b*128 + c)*N + n]
                        : Bv[((long long)b*128 + (c-128))*N + n];
    out[t] = v;
}

__global__ void gather_qk(const float* __restrict__ q, const float* __restrict__ k,
                          const int* __restrict__ idx, float* __restrict__ out, int N, int K)
{   // out[b,c,n,kk] = q[b,c,n] - k[b,c,idx[b,n,kk]]
    long long t = (long long)blockIdx.x * 256 + threadIdx.x;
    long long total = (long long)8 * 64 * N * K;
    if (t >= total) return;
    int kk = (int)(t % K);
    int n  = (int)((t / K) % N);
    int c  = (int)((t / ((long long)K * N)) % 64);
    int b  = (int)(t / ((long long)64 * K * N));
    int iv = idx[((long long)b*N + n)*K + kk];
    out[t] = q[((long long)b*64 + c)*N + n] - k[((long long)b*64 + c)*N + iv];
}

__global__ void gather_pos(const float* __restrict__ pos, const int* __restrict__ idx,
                           float* __restrict__ out, int N, int K)
{   // out[b,c,n,kk] = pos[b,c,n] - pos[b,c,idx[b,n,kk]],  c < 3
    long long t = (long long)blockIdx.x * 256 + threadIdx.x;
    long long total = (long long)8 * 3 * N * K;
    if (t >= total) return;
    int kk = (int)(t % K);
    int n  = (int)((t / K) % N);
    int c  = (int)((t / ((long long)K * N)) % 3);
    int b  = (int)(t / ((long long)3 * K * N));
    int iv = idx[((long long)b*N + n)*K + kk];
    out[t] = pos[((long long)b*3 + c)*N + n] - pos[((long long)b*3 + c)*N + iv];
}

__global__ void add_inplace(float* __restrict__ dst, const float* __restrict__ src,
                            long long count)
{
    long long t = (long long)blockIdx.x * 256 + threadIdx.x;
    if (t < count) dst[t] += src[t];
}

__global__ void softmax_agg(const float* __restrict__ a, const float* __restrict__ v,
                            const float* __restrict__ pe, float* __restrict__ agg, int N)
{   // t == (b*64+c)*N + n ; softmax over K=16 then sum a*(v+pe)
    long long t = (long long)blockIdx.x * 256 + threadIdx.x;
    long long total = (long long)8 * 64 * N;
    if (t >= total) return;
    const float* ap = a + t * 16;
    float lg[16];
    float mx = -3.4e38f;
#pragma unroll
    for (int kk = 0; kk < 16; ++kk) { lg[kk] = ap[kk]; mx = fmaxf(mx, lg[kk]); }
    float s = 0.f;
#pragma unroll
    for (int kk = 0; kk < 16; ++kk) { float e = __expf(lg[kk] - mx); lg[kk] = e; s += e; }
    float vn = v[t];
    const float* pp = pe + t * 16;
    float acc = 0.f;
#pragma unroll
    for (int kk = 0; kk < 16; ++kk) acc += lg[kk] * (vn + pp[kk]);
    agg[t] = acc / s;
}

__global__ void hup_fill(const float* __restrict__ H, float* __restrict__ dfin, int N2)
{   // df_in rows 128..255 <- repeat(H, 2)
    long long t = (long long)blockIdx.x * 256 + threadIdx.x;
    long long total = (long long)8 * 128 * N2;
    if (t >= total) return;
    int col = (int)(t % N2);
    int c   = (int)((t / N2) % 128);
    int b   = (int)(t / ((long long)128 * N2));
    dfin[((long long)b*256 + 128 + c)*N2 + col] =
        H[((long long)b*128 + c)*(N2/2) + (col >> 1)];
}

__global__ void pst_transpose(const float* __restrict__ w, float* __restrict__ wt)
{   // wt[j][o][c] = pst_w[c][o][j] ; pst_w shape (32,128,2)
    int t = blockIdx.x * 256 + threadIdx.x;
    if (t >= 2*128*32) return;
    int c = t % 32;
    int o = (t / 32) % 128;
    int j = t / 4096;
    wt[t] = w[(c*128 + o)*2 + j];
}

// ---------------------------------------------------------------------------
// Host orchestration
// ---------------------------------------------------------------------------
extern "C" void kernel_launch(void* const* d_in, const int* in_sizes, int n_in,
                              void* d_out, int out_size, void* d_ws, size_t ws_size,
                              hipStream_t stream)
{
    (void)in_sizes; (void)n_in; (void)out_size; (void)ws_size;
    const int Bn = 8, N = 2048, Kn = 16, N2 = 4096;   // N2 = N*UP
    const int NK = N * Kn;                             // 32768

    const float* pcd   = (const float*)d_in[0];
    const float* fg    = (const float*)d_in[1];
    const float* Kprev = (const float*)d_in[2];
    // params in _make_params insertion order
    const float* m1_w1 = (const float*)d_in[3];  const float* m1_b1 = (const float*)d_in[4];
    const float* m1_w2 = (const float*)d_in[5];  const float* m1_b2 = (const float*)d_in[6];
    const float* m2_w1 = (const float*)d_in[7];  const float* m2_b1 = (const float*)d_in[8];
    const float* m2_w2 = (const float*)d_in[9];  const float* m2_b2 = (const float*)d_in[10];
    const float* mv_w1 = (const float*)d_in[11]; const float* mv_b1 = (const float*)d_in[12];
    const float* mv_w2 = (const float*)d_in[13]; const float* mv_b2 = (const float*)d_in[14];
    const float* mv_ws = (const float*)d_in[15]; const float* mv_bs = (const float*)d_in[16];
    const float* st_wk = (const float*)d_in[17]; const float* st_bk = (const float*)d_in[18];
    const float* st_wq = (const float*)d_in[19]; const float* st_bq = (const float*)d_in[20];
    const float* st_wv = (const float*)d_in[21]; const float* st_bv = (const float*)d_in[22];
    const float* pos_w1 = (const float*)d_in[23]; const float* pos_b1 = (const float*)d_in[24];
    const float* pos_g  = (const float*)d_in[25]; const float* pos_be = (const float*)d_in[26];
    const float* pos_m  = (const float*)d_in[27]; const float* pos_v  = (const float*)d_in[28];
    const float* pos_w2 = (const float*)d_in[29]; const float* pos_b2 = (const float*)d_in[30];
    const float* at_w1 = (const float*)d_in[31]; const float* at_b1 = (const float*)d_in[32];
    const float* at_g  = (const float*)d_in[33]; const float* at_be = (const float*)d_in[34];
    const float* at_m  = (const float*)d_in[35]; const float* at_v  = (const float*)d_in[36];
    const float* at_w2 = (const float*)d_in[37]; const float* at_b2 = (const float*)d_in[38];
    const float* st_we = (const float*)d_in[39]; const float* st_be_ = (const float*)d_in[40];
    const float* ps_w1 = (const float*)d_in[41]; const float* ps_b1 = (const float*)d_in[42];
    const float* ps_w2 = (const float*)d_in[43]; const float* ps_b2 = (const float*)d_in[44];
    const float* pst_w = (const float*)d_in[45];
    const float* df_w1 = (const float*)d_in[46]; const float* df_b1 = (const float*)d_in[47];
    const float* df_w2 = (const float*)d_in[48]; const float* df_b2 = (const float*)d_in[49];
    const float* df_ws = (const float*)d_in[50]; const float* df_bs = (const float*)d_in[51];
    const float* d_w1  = (const float*)d_in[52]; const float* d_b1  = (const float*)d_in[53];
    const float* d_w2  = (const float*)d_in[54]; const float* d_b2  = (const float*)d_in[55];

    // workspace layout (floats)
    float* ws = (float*)d_ws;
    size_t off = 0;
    auto alloc = [&](size_t n) { float* p = ws + off; off += n; return p; };
    float* f1a   = alloc((size_t)64*N*Bn);
    float* f1cat = alloc((size_t)768*N*Bn);
    float* gmaxb = alloc((size_t)128*Bn);
    float* q2a   = alloc((size_t)256*N*Bn);
    float* Qb    = alloc((size_t)128*N*Bn);
    float* catKQ = alloc((size_t)256*N*Bn);
    float* hv    = alloc((size_t)128*N*Bn);
    float* scv   = alloc((size_t)128*N*Bn);
    float* valb  = alloc((size_t)128*N*Bn);
    float* kbuf  = alloc((size_t)64*N*Bn);
    float* qbuf  = alloc((size_t)64*N*Bn);
    float* vbuf  = alloc((size_t)64*N*Bn);
    int*   idxb  = (int*)alloc((size_t)N*Kn*Bn);
    float* qkrel = alloc((size_t)64*NK*Bn);      // later: attn input (qk_rel + pe)
    float* pe1   = alloc((size_t)64*NK*Bn);      // later reused for attn logits
    float* peb   = alloc((size_t)64*NK*Bn);
    float* posrel= alloc((size_t)3*NK*Bn);
    float* aggb  = alloc((size_t)64*N*Bn);
    float* Hb    = alloc((size_t)128*N*Bn);
    float* psf   = alloc((size_t)64*N*Bn);
    float* fcb   = alloc((size_t)32*N*Bn);
    float* wjt   = alloc((size_t)2*128*32);
    float* dfin  = alloc((size_t)256*N2*Bn);
    float* dfh   = alloc((size_t)128*N2*Bn);
    float* dfsc  = alloc((size_t)128*N2*Bn);
    float* d1b   = alloc((size_t)64*N2*Bn);

    float* pcd_child = (float*)d_out;                         // (8,3,4096)
    float* Kcur      = (float*)d_out + (size_t)Bn*3*N2;       // (8,128,4096)

    auto gemm = [&](const float* W, const float* X, const float* bias,
                    const float* res, float* Y,
                    int O, int C, int M,
                    long long sX, int lX, long long sY, int lY,
                    long long sR, int lR, int cs, int co, int flags,
                    const float* g, const float* be, const float* bm, const float* bv)
    {
        dim3 grid((M + 63) / 64, (O + 31) / 32, Bn);
        wmma_gemm<<<grid, 256, 0, stream>>>(W, X, bias, g, be, bm, bv, res, Y,
                                            O, C, M, sX, lX, sY, lY, sR, lR,
                                            cs, co, flags);
    };
    const float* nf = nullptr;

    // ---- feature extractor mlp_1 ----
    gemm(m1_w1, pcd, m1_b1, nf, f1a, 64, 3, N, 3LL*N, N, 64LL*N, N, 0, 0, 1, 0,
         F_RELU_OUT, nf, nf, nf, nf);
    gemm(m1_w2, f1a, m1_b2, nf, f1cat, 128, 64, N, 64LL*N, N, 768LL*N, N, 0, 0, 1, 0,
         0, nf, nf, nf, nf);
    rowmax_kernel<<<1024, 256, 0, stream>>>(f1cat, gmaxb, N);
    {
        long long total = (long long)Bn * 640 * N;
        fill_f1cat<<<(unsigned)((total + 255) / 256), 256, 0, stream>>>(f1cat, gmaxb, fg, N);
    }
    // ---- mlp_2 -> Q ----
    gemm(m2_w1, f1cat, m2_b1, nf, q2a, 256, 768, N, 768LL*N, N, 256LL*N, N, 0, 0, 1, 0,
         F_RELU_OUT, nf, nf, nf, nf);
    gemm(m2_w2, q2a, m2_b2, nf, Qb, 128, 256, N, 256LL*N, N, 128LL*N, N, 0, 0, 1, 0,
         0, nf, nf, nf, nf);

    // ---- skip transformer: value = mlp_res(cat(K_prev, Q)) ----
    {
        long long total = (long long)Bn * 256 * N;
        concat128<<<(unsigned)((total + 255) / 256), 256, 0, stream>>>(Kprev, Qb, catKQ, N);
    }
    gemm(mv_ws, catKQ, mv_bs, nf, scv, 128, 256, N, 256LL*N, N, 128LL*N, N, 0, 0, 1, 0,
         0, nf, nf, nf, nf);
    gemm(mv_w1, catKQ, mv_b1, nf, hv, 128, 256, N, 256LL*N, N, 128LL*N, N, 0, 0, 1, 0,
         F_RELU_OUT, nf, nf, nf, nf);
    gemm(mv_w2, hv, mv_b2, scv, valb, 128, 128, N, 128LL*N, N, 128LL*N, N, 128LL*N, N, 1, 0,
         F_RES, nf, nf, nf, nf);
    // k, q, v projections
    gemm(st_wk, Kprev, st_bk, nf, kbuf, 64, 128, N, 128LL*N, N, 64LL*N, N, 0, 0, 1, 0,
         0, nf, nf, nf, nf);
    gemm(st_wq, Qb, st_bq, nf, qbuf, 64, 128, N, 128LL*N, N, 64LL*N, N, 0, 0, 1, 0,
         0, nf, nf, nf, nf);
    gemm(st_wv, valb, st_bv, nf, vbuf, 64, 128, N, 128LL*N, N, 64LL*N, N, 0, 0, 1, 0,
         0, nf, nf, nf, nf);
    // knn + gathers
    knn_kernel<<<dim3(N / 256, Bn), 256, 0, stream>>>(pcd, idxb, N);
    {
        long long total = (long long)Bn * 64 * N * Kn;
        gather_qk<<<(unsigned)((total + 255) / 256), 256, 0, stream>>>(qbuf, kbuf, idxb,
                                                                       qkrel, N, Kn);
        long long total3 = (long long)Bn * 3 * N * Kn;
        gather_pos<<<(unsigned)((total3 + 255) / 256), 256, 0, stream>>>(pcd, idxb,
                                                                         posrel, N, Kn);
    }
    // positional encoding: conv(3->64)+BN+relu, conv(64->64)
    gemm(pos_w1, posrel, pos_b1, nf, pe1, 64, 3, NK, 3LL*NK, NK, 64LL*NK, NK, 0, 0, 1, 0,
         F_BN | F_RELU_OUT, pos_g, pos_be, pos_m, pos_v);
    gemm(pos_w2, pe1, pos_b2, nf, peb, 64, 64, NK, 64LL*NK, NK, 64LL*NK, NK, 0, 0, 1, 0,
         0, nf, nf, nf, nf);
    // attn logits = fused(W2 @ relu(BN(W1 @ (qk_rel + pe)))) ; reuse pe1 buffer
    add_inplace<<<(unsigned)(((long long)Bn*64*NK + 255) / 256), 256, 0, stream>>>(
        qkrel, peb, (long long)Bn * 64 * NK);
    attn_fused<<<dim3(NK / 64, Bn), 256, 0, stream>>>(at_w1, at_b1, at_g, at_be, at_m, at_v,
                                                      at_w2, at_b2, qkrel, pe1, NK);
    // softmax over K + aggregate with (v + pe)
    softmax_agg<<<(unsigned)(((long long)Bn*64*N + 255) / 256), 256, 0, stream>>>(
        pe1, vbuf, peb, aggb, N);
    // H = conv(agg) + value
    gemm(st_we, aggb, st_be_, valb, Hb, 128, 64, N, 64LL*N, N, 128LL*N, N, 128LL*N, N, 1, 0,
         F_RES, nf, nf, nf, nf);

    // ---- point-shuffle branch ----
    gemm(ps_w1, Hb, ps_b1, nf, psf, 64, 128, N, 128LL*N, N, 64LL*N, N, 0, 0, 1, 0,
         F_RELU_OUT, nf, nf, nf, nf);
    gemm(ps_w2, psf, ps_b2, nf, fcb, 32, 64, N, 64LL*N, N, 32LL*N, N, 0, 0, 1, 0,
         0, nf, nf, nf, nf);
    pst_transpose<<<32, 256, 0, stream>>>(pst_w, wjt);
    // ConvTranspose1d as two scattered GEMMs (ycol = 2*col + j)
    gemm(wjt, fcb, nf, nf, dfin, 128, 32, N, 32LL*N, N, 256LL*N2, N2, 0, 0, 2, 0,
         0, nf, nf, nf, nf);
    gemm(wjt + 4096, fcb, nf, nf, dfin, 128, 32, N, 32LL*N, N, 256LL*N2, N2, 0, 0, 2, 1,
         0, nf, nf, nf, nf);
    hup_fill<<<(unsigned)(((long long)Bn*128*N2 + 255) / 256), 256, 0, stream>>>(Hb, dfin, N2);

    // ---- K_curr = mlp_res(cat(fc2, H_up)) -> directly into d_out ----
    gemm(df_ws, dfin, df_bs, nf, dfsc, 128, 256, N2, 256LL*N2, N2, 128LL*N2, N2, 0, 0, 1, 0,
         0, nf, nf, nf, nf);
    gemm(df_w1, dfin, df_b1, nf, dfh, 128, 256, N2, 256LL*N2, N2, 128LL*N2, N2, 0, 0, 1, 0,
         F_RELU_OUT, nf, nf, nf, nf);
    gemm(df_w2, dfh, df_b2, dfsc, Kcur, 128, 128, N2, 128LL*N2, N2, 128LL*N2, N2,
         128LL*N2, N2, 1, 0, F_RES, nf, nf, nf, nf);

    // ---- delta head: relu(K_curr) -> conv+relu -> conv -> tanh ; +repeat(pcd) ----
    gemm(d_w1, Kcur, d_b1, nf, d1b, 64, 128, N2, 128LL*N2, N2, 64LL*N2, N2, 0, 0, 1, 0,
         F_RELU_IN | F_RELU_OUT, nf, nf, nf, nf);
    gemm(d_w2, d1b, d_b2, pcd, pcd_child, 3, 64, N2, 64LL*N2, N2, 3LL*N2, N2, 3LL*N, N, 1, 0,
         F_TANH | F_RES | F_RES_HALF, nf, nf, nf, nf);
}
